// GraphConvpoolMPNN_block_49795850830069
// MI455X (gfx1250) — matrix operand order repaired and checked by reading, not compiled, not run on hardware
//
#include <hip/hip_runtime.h>
#include <hip/hip_bf16.h>

// MI455X / gfx1250, wave32. All matmuls via v_wmma_f32_16x16x32_bf16.
// Attention staging uses GLOBAL_LOAD_ASYNC_TO_LDS_B128 (ASYNCcnt) when the
// toolchain exposes the builtin, with a plain LDS-copy fallback.

typedef __attribute__((ext_vector_type(16))) __bf16 v16bf;
typedef __attribute__((ext_vector_type(8)))  float  v8f;

#define DEVFN static __device__ __forceinline__

#if __has_builtin(__builtin_amdgcn_global_load_async_to_lds_b128)
#define HAS_ASYNC_LDS 1
#else
#define HAS_ASYNC_LDS 0
#endif

struct B32x8 { unsigned int u[8]; };

// Two contiguous 16-byte chunks of bf16 -> one 16-element WMMA operand.
DEVFN v16bf ld16x2(const __bf16* p0, const __bf16* p1) {
  uint4 x = *(const uint4*)p0;
  uint4 y = *(const uint4*)p1;
  B32x8 t;
  t.u[0]=x.x; t.u[1]=x.y; t.u[2]=x.z; t.u[3]=x.w;
  t.u[4]=y.x; t.u[5]=y.y; t.u[6]=y.z; t.u[7]=y.w;
  return __builtin_bit_cast(v16bf, t);
}

DEVFN v16bf pack16(const float* f) {
  v16bf r;
#pragma unroll
  for (int i = 0; i < 16; ++i) r[i] = (__bf16)f[i];
  return r;
}

DEVFN v8f wmma_bf(v16bf a, v16bf b, v8f c) {
  return __builtin_amdgcn_wmma_f32_16x16x32_bf16(false, a, false, b, (short)0, c,
                                                 false, false);
}

DEVFN v8f vzero() { v8f z = {0.f,0.f,0.f,0.f,0.f,0.f,0.f,0.f}; return z; }

// ---------------------------------------------------------------------------
// Kernel 0: zero the stats scratch (bn0 + bn1 sum/sumsq + derived g/k).
__global__ void k_zero(float* p) {
  int i = threadIdx.x;
  if (i < 512) p[i] = 0.f;
}

// ---------------------------------------------------------------------------
// Convert the three 64x64 fp32 weight matrices to bf16 once (saves per-wave
// v_cvt_pk_bf16_f32 work in the GEMM kernels).
__global__ void k_prep(const float* __restrict__ w0, const float* __restrict__ w1,
                       const float* __restrict__ w2, __bf16* __restrict__ wb) {
  int idx = blockIdx.x * 256 + threadIdx.x;     // 3 * 4096
  int which = idx >> 12, off = idx & 4095;
  const float* src = (which == 0) ? w0 : (which == 1) ? w1 : w2;
  wb[idx] = (__bf16)src[off];
}

// ---------------------------------------------------------------------------
// Kernel 1: nf = X @ Wmap^T + b  (X gathered from windowed x), bf16 out,
// plus bn0 sum/sumsq accumulation. One wave per 16-row tile.
__global__ void __launch_bounds__(128) k_nf(
    const float* __restrict__ x, const __bf16* __restrict__ wmapb,
    const float* __restrict__ bmap, __bf16* __restrict__ nf,
    float* __restrict__ stats) {
  __shared__ float sred[128];
  int tid = threadIdx.x;
  if (tid < 128) sred[tid] = 0.f;
  __syncthreads();
  int wave = tid >> 5, lane = tid & 31, l16 = lane & 15, hh = lane >> 4;
  int bb = blockIdx.x >> 3;                 // batch index b' in [0,256)
  int rt = ((blockIdx.x & 7) << 2) + wave;  // row tile [0,32)
  int rowbase = rt << 4;
  int b = bb >> 4, l = bb & 15;
  int m = rowbase >> 8;                     // moving-window slot
  int n = (rowbase & 255) + l16;
  const float* xrow = x + (((b * 17 + l + m) * 256 + n) * 64);
  v16bf A[2];
#pragma unroll
  for (int kc = 0; kc < 2; ++kc) {          // K = 64 in two chunks of 32
    int kb = kc * 32;
    float t[16];
    float4 q0 = *(const float4*)(xrow + kb + hh*8);
    float4 q1 = *(const float4*)(xrow + kb + hh*8 + 4);
    float4 q2 = *(const float4*)(xrow + kb + 16 + hh*8);
    float4 q3 = *(const float4*)(xrow + kb + 16 + hh*8 + 4);
    t[0]=q0.x; t[1]=q0.y; t[2]=q0.z;  t[3]=q0.w;
    t[4]=q1.x; t[5]=q1.y; t[6]=q1.z;  t[7]=q1.w;
    t[8]=q2.x; t[9]=q2.y; t[10]=q2.z; t[11]=q2.w;
    t[12]=q3.x; t[13]=q3.y; t[14]=q3.z; t[15]=q3.w;
    A[kc] = pack16(t);
  }
#pragma unroll
  for (int ot = 0; ot < 4; ++ot) {
    int c = ot * 16 + l16;
    v8f acc = vzero();
#pragma unroll
    for (int kc = 0; kc < 2; ++kc) {
      const __bf16* wr = wmapb + c * 64 + kc * 32 + hh * 8;
      acc = wmma_bf(A[kc], ld16x2(wr, wr + 16), acc);
    }
    float bias = bmap[c];
    float ps = 0.f, pq = 0.f;
#pragma unroll
    for (int r = 0; r < 8; ++r) {
      float v = acc[r] + bias;
      int row = rowbase + r + 8 * hh;
      nf[(bb * 512 + row) * 64 + c] = (__bf16)v;
      ps += v; pq += v * v;
    }
    atomicAdd(&sred[c], ps);
    atomicAdd(&sred[64 + c], pq);
  }
  __syncthreads();
  if (tid < 128) atomicAdd(&stats[tid], sred[tid]);
}

// ---------------------------------------------------------------------------
// Finalize BN stats: g = w * rsqrt(var+eps), k = b - mean*g.
__global__ void k_bnfin(const float* __restrict__ s, const float* __restrict__ w,
                        const float* __restrict__ b, float* __restrict__ g,
                        float* __restrict__ k) {
  int c = threadIdx.x;
  if (c < 64) {
    const float n = 131072.f;  // 256 * 512
    float mean = s[c] / n;
    float var = s[64 + c] / n - mean * mean;
    float gg = w[c] * rsqrtf(var + 1e-5f);
    g[c] = gg;
    k[c] = b[c] - mean * gg;
  }
}

// ---------------------------------------------------------------------------
// xbT[b][c][i] = bn0(nf[b][i][c]) as bf16, transposed for WMMA B-tiles.
__global__ void k_xbT(const __bf16* __restrict__ nf, const float* __restrict__ g,
                      const float* __restrict__ k, __bf16* __restrict__ xbT) {
  int idx = blockIdx.x * 256 + threadIdx.x;   // 256*64*512 total
  int i = idx & 511;
  int c = (idx >> 9) & 63;
  int bbq = idx >> 15;
  float v = (float)nf[(bbq * 512 + i) * 64 + c] * g[c] + k[c];
  xbT[idx] = (__bf16)v;
}

// ---------------------------------------------------------------------------
// Cooperative staging of one 32-column j-chunk: 32x64 bf16 of nf (score B)
// and 64x32 bf16 of vT (value B). 256 threads, one b128 each per buffer.
DEVFN void stage_chunk(const __bf16* nfb, const __bf16* vTb, int jbase,
                       __bf16* snf, __bf16* svT, int tid) {
  int rrow = tid >> 3, rseg = (tid & 7) * 8;
  const __bf16* gs = nfb + (jbase + rrow) * 64 + rseg;
  __bf16* ls = snf + rrow * 64 + rseg;
  int vc = tid >> 2, vseg = (tid & 3) * 8;
  const __bf16* gv = vTb + vc * 512 + jbase + vseg;
  __bf16* lv = svT + vc * 32 + vseg;
#if HAS_ASYNC_LDS
  // Builtin signature (from hipcc diagnostic): pointee is GCC-vector
  // 'int __attribute__((vector_size(16)))', global (AS1) src, local (AS3) dst.
  typedef int v4i_gcc __attribute__((vector_size(16)));
  typedef __attribute__((address_space(1))) v4i_gcc as1_v4i;
  typedef __attribute__((address_space(3))) v4i_gcc as3_v4i;
  __builtin_amdgcn_global_load_async_to_lds_b128((as1_v4i*)gs, (as3_v4i*)ls, 0, 0);
  __builtin_amdgcn_global_load_async_to_lds_b128((as1_v4i*)gv, (as3_v4i*)lv, 0, 0);
#else
  *(uint4*)ls = *(const uint4*)gs;
  *(uint4*)lv = *(const uint4*)gv;
#endif
}

DEVFN void wait_stage() {
#if HAS_ASYNC_LDS
  asm volatile("s_wait_asynccnt 0x0" ::: "memory");
#endif
}

// ---------------------------------------------------------------------------
// Attention hop: out_i = sum_j mask_ij * softmax_j(leaky(nf_i.nf_j - diag)) v_j
//                + v_i   (the "+eye" term; mask diag == 1)
// Flash-style online softmax; one workgroup (8 waves) per batch; each wave
// owns 4 row tiles of 16; 32-column j-chunks double-buffered in LDS via
// async global->LDS loads. P-tile transposed D->A layout via LDS scratch.
__global__ void __launch_bounds__(256) k_attn(
    const __bf16* __restrict__ nf, const __bf16* __restrict__ vT,
    __bf16* __restrict__ outRM, __bf16* __restrict__ outT, int writeT) {
  __shared__ __bf16 snf[2][32 * 64];          // score-B stage, 8 KB
  __shared__ __bf16 svT[2][64 * 32];          // value-B stage, 8 KB
  __shared__ __bf16 pscr[8][512];             // 16x32 P-tile per wave, 8 KB
  int tid = threadIdx.x;
  int wave = tid >> 5, lane = tid & 31, l16 = lane & 15, hh = lane >> 4;
  int bb = blockIdx.x;
  const __bf16* nfb = nf + bb * (512 * 64);
  const __bf16* vTb = vT + bb * (64 * 512);
  __bf16* ps = &pscr[wave][0];
  for (int it = 0; it < 4; ++it) {
    int rowbase = ((it << 3) + wave) << 4;
    const __bf16* ar = nfb + (rowbase + l16) * 64;
    v16bf A[2];
#pragma unroll
    for (int kc = 0; kc < 2; ++kc)
      A[kc] = ld16x2(ar + kc * 32 + hh * 8, ar + kc * 32 + 16 + hh * 8);
    float m_[8], s_[8];
    v8f o_[4];
#pragma unroll
    for (int r = 0; r < 8; ++r) { m_[r] = -3.0e38f; s_[r] = 0.f; }
#pragma unroll
    for (int t2 = 0; t2 < 4; ++t2) o_[t2] = vzero();
    int mrow = rowbase >> 8;
    // preload chunk 0
    stage_chunk(nfb, vTb, 0, snf[0], svT[0], tid);
    wait_stage();
    __syncthreads();
    for (int jc = 0; jc < 16; ++jc) {
      int jbase = jc << 5;
      int cur = jc & 1;
      if (jc + 1 < 16)
        stage_chunk(nfb, vTb, jbase + 32, snf[1 - cur], svT[1 - cur], tid);
      const __bf16* snf_c = snf[cur];
      const __bf16* svT_c = svT[cur];
      // ---- scores S = nf_rows (16x64) @ nf_cols^T (64x32), 4 WMMAs ----
      v8f S[2];
#pragma unroll
      for (int js = 0; js < 2; ++js) {
        v8f sacc = vzero();
#pragma unroll
        for (int kc = 0; kc < 2; ++kc) {
          const __bf16* br = snf_c + (js * 16 + l16) * 64 + kc * 32 + hh * 8;
          sacc = wmma_bf(A[kc], ld16x2(br, br + 16), sacc);
        }
        S[js] = sacc;
      }
      // ---- leaky_relu(S - 1e8*eye) ----
#pragma unroll
      for (int js = 0; js < 2; ++js)
#pragma unroll
        for (int r = 0; r < 8; ++r) {
          int row = rowbase + r + 8 * hh;
          int col = jbase + js * 16 + l16;
          float z = S[js][r];
          if (row == col) z -= 1.0e8f;
          S[js][r] = (z > 0.f) ? z : 0.01f * z;
        }
      // ---- online softmax: row max across 16 lanes of each half ----
      float alpha[8];
#pragma unroll
      for (int r = 0; r < 8; ++r) {
        float mx = fmaxf(S[0][r], S[1][r]);
        mx = fmaxf(mx, __shfl_xor(mx, 1, 32));
        mx = fmaxf(mx, __shfl_xor(mx, 2, 32));
        mx = fmaxf(mx, __shfl_xor(mx, 4, 32));
        mx = fmaxf(mx, __shfl_xor(mx, 8, 32));
        float mn = fmaxf(m_[r], mx);
        alpha[r] = __expf(m_[r] - mn);
        m_[r] = mn;
      }
#pragma unroll
      for (int js = 0; js < 2; ++js)
#pragma unroll
        for (int r = 0; r < 8; ++r) S[js][r] = __expf(S[js][r] - m_[r]);
#pragma unroll
      for (int r = 0; r < 8; ++r) {
        float sm = S[0][r] + S[1][r];
        sm += __shfl_xor(sm, 1, 32);
        sm += __shfl_xor(sm, 2, 32);
        sm += __shfl_xor(sm, 4, 32);
        sm += __shfl_xor(sm, 8, 32);
        s_[r] = s_[r] * alpha[r] + sm;
      }
#pragma unroll
      for (int t2 = 0; t2 < 4; ++t2)
#pragma unroll
        for (int r = 0; r < 8; ++r) o_[t2][r] = o_[t2][r] * alpha[r];
      // ---- decay mask is block-constant; fold into numerator ----
      float mask = (mrow == (jbase >> 8)) ? 1.0f : 0.7f;
#pragma unroll
      for (int js = 0; js < 2; ++js)
#pragma unroll
        for (int r = 0; r < 8; ++r) {
          int prow = r + 8 * hh;
          int pcol = js * 16 + l16;
          ps[prow * 32 + pcol] = (__bf16)(S[js][r] * mask);
        }
      asm volatile("s_wait_dscnt 0x0" ::: "memory");
      // ---- reload P in A-matrix layout; accumulate o += P @ V chunk ----
      const __bf16* pr = ps + l16 * 32;
      v16bf Ap = ld16x2(pr + hh * 8, pr + 16 + hh * 8);
#pragma unroll
      for (int ct = 0; ct < 4; ++ct) {
        const __bf16* vr = svT_c + (ct * 16 + l16) * 32 + hh * 8;
        o_[ct] = wmma_bf(Ap, ld16x2(vr, vr + 16), o_[ct]);
      }
      asm volatile("s_wait_dscnt 0x0" ::: "memory");
      // all waves done reading buffer `cur`; next buffer complete after wait
      if (jc + 1 < 16) {
        wait_stage();
        __syncthreads();
      }
    }
    // ---- normalize, add diagonal (eye) term, store ----
#pragma unroll
    for (int ct = 0; ct < 4; ++ct) {
      int c = ct * 16 + l16;
#pragma unroll
      for (int r = 0; r < 8; ++r) {
        int row = rowbase + r + 8 * hh;
        float val = o_[ct][r] / s_[r] + (float)vTb[c * 512 + row];
        outRM[(bb * 512 + row) * 64 + c] = (__bf16)val;
        if (writeT) outT[bb * (64 * 512) + c * 512 + row] = (__bf16)val;
      }
    }
    __syncthreads();   // protect stage buffers before next row-tile restages
  }
}

// ---------------------------------------------------------------------------
// out_pre = h1 @ th0^T + h2 @ th1^T + b0 + b1 (fp32), plus bn1 stats.
__global__ void __launch_bounds__(128) k_proj(
    const __bf16* __restrict__ h1, const __bf16* __restrict__ h2,
    const __bf16* __restrict__ th0wb, const float* __restrict__ th0b,
    const __bf16* __restrict__ th1wb, const float* __restrict__ th1b,
    float* __restrict__ outpre, float* __restrict__ stats) {
  __shared__ float sred[128];
  int tid = threadIdx.x;
  if (tid < 128) sred[tid] = 0.f;
  __syncthreads();
  int wave = tid >> 5, lane = tid & 31, l16 = lane & 15, hh = lane >> 4;
  int bb = blockIdx.x >> 3;
  int rt = ((blockIdx.x & 7) << 2) + wave;
  int rowbase = rt << 4;
  const __bf16* a1r = h1 + (bb * 512 + rowbase + l16) * 64;
  const __bf16* a2r = h2 + (bb * 512 + rowbase + l16) * 64;
  v16bf A1[2], A2[2];
#pragma unroll
  for (int kc = 0; kc < 2; ++kc) {
    A1[kc] = ld16x2(a1r + kc * 32 + hh * 8, a1r + kc * 32 + 16 + hh * 8);
    A2[kc] = ld16x2(a2r + kc * 32 + hh * 8, a2r + kc * 32 + 16 + hh * 8);
  }
#pragma unroll
  for (int ot = 0; ot < 4; ++ot) {
    int c = ot * 16 + l16;
    v8f acc = vzero();
#pragma unroll
    for (int kc = 0; kc < 2; ++kc) {
      const __bf16* w0 = th0wb + c * 64 + kc * 32 + hh * 8;
      const __bf16* w1 = th1wb + c * 64 + kc * 32 + hh * 8;
      acc = wmma_bf(A1[kc], ld16x2(w0, w0 + 16), acc);
      acc = wmma_bf(A2[kc], ld16x2(w1, w1 + 16), acc);
    }
    float bias = th0b[c] + th1b[c];
    float psum = 0.f, pq = 0.f;
#pragma unroll
    for (int r = 0; r < 8; ++r) {
      float v = acc[r] + bias;
      int row = rowbase + r + 8 * hh;
      outpre[(bb * 512 + row) * 64 + c] = v;
      psum += v; pq += v * v;
    }
    atomicAdd(&sred[c], psum);
    atomicAdd(&sred[64 + c], pq);
  }
  __syncthreads();
  if (tid < 128) atomicAdd(&stats[tid], sred[tid]);
}

// ---------------------------------------------------------------------------
// Final: leaky(bn1(out_pre)) then mean over moving-window (2) and windows (16).
__global__ void k_final(const float* __restrict__ outpre,
                        const float* __restrict__ g, const float* __restrict__ k,
                        float* __restrict__ out) {
  int idx = blockIdx.x * 256 + threadIdx.x;   // 16*256*64 outputs
  int c = idx & 63;
  int n = (idx >> 6) & 255;
  int b = idx >> 14;
  float gg = g[c], kk = k[c], acc = 0.f;
#pragma unroll 4
  for (int l = 0; l < 16; ++l)
    for (int m = 0; m < 2; ++m) {
      float v = outpre[(((b * 16 + l) * 512) + m * 256 + n) * 64 + c];
      v = gg * v + kk;
      acc += (v > 0.f) ? v : 0.01f * v;
    }
  out[idx] = acc * (1.0f / 32.0f);
}

// ---------------------------------------------------------------------------
extern "C" void kernel_launch(void* const* d_in, const int* in_sizes, int n_in,
                              void* d_out, int out_size, void* d_ws, size_t ws_size,
                              hipStream_t stream) {
  const float* x    = (const float*)d_in[0];
  const float* wmap = (const float*)d_in[1];
  const float* bmap = (const float*)d_in[2];
  const float* bn0w = (const float*)d_in[3];
  const float* bn0b = (const float*)d_in[4];
  const float* th0w = (const float*)d_in[5];
  const float* th0b = (const float*)d_in[6];
  const float* th1w = (const float*)d_in[7];
  const float* th1b = (const float*)d_in[8];
  const float* bn1w = (const float*)d_in[9];
  const float* bn1b = (const float*)d_in[10];
  float* out = (float*)d_out;

  const size_t SZ = 256ull * 512 * 64 * 2;   // one [256,512,64] bf16 buffer
  char* ws = (char*)d_ws;
  __bf16* nf     = (__bf16*)(ws + 0 * SZ);
  __bf16* xbT    = (__bf16*)(ws + 1 * SZ);
  __bf16* h1     = (__bf16*)(ws + 2 * SZ);
  __bf16* h1T    = (__bf16*)(ws + 3 * SZ);
  __bf16* h2     = (__bf16*)(ws + 4 * SZ);
  float*  outpre = (float*)(ws + 5 * SZ);    // fp32: occupies 2*SZ
  float*  stats  = (float*)(ws + 7 * SZ);    // 512 floats (2 KB)
  __bf16* wb     = (__bf16*)(ws + 7 * SZ + 4096);  // 3 x 64x64 bf16 weights

  k_zero<<<1, 512, 0, stream>>>(stats);
  k_prep<<<dim3(48), dim3(256), 0, stream>>>(wmap, th0w, th1w, wb);
  k_nf<<<dim3(2048), dim3(128), 0, stream>>>(x, wb, bmap, nf, stats);
  k_bnfin<<<1, 64, 0, stream>>>(stats, bn0w, bn0b, stats + 128, stats + 192);
  k_xbT<<<dim3(32768), dim3(256), 0, stream>>>(nf, stats + 128, stats + 192, xbT);
  k_attn<<<dim3(256), dim3(256), 0, stream>>>(nf, xbT, h1, h1T, 1);   // hop 1
  k_attn<<<dim3(256), dim3(256), 0, stream>>>(nf, h1T, h2, h1T, 0);   // hop 2
  k_proj<<<dim3(2048), dim3(128), 0, stream>>>(h1, h2, wb + 4096, th0b,
                                               wb + 8192, th1b, outpre, stats + 256);
  k_bnfin<<<1, 64, 0, stream>>>(stats + 256, bn1w, bn1b, stats + 384, stats + 448);
  k_final<<<dim3(1024), dim3(256), 0, stream>>>(outpre, stats + 384, stats + 448, out);

  (void)in_sizes; (void)n_in; (void)out_size; (void)ws_size;
}
